// NewLLMResidual_41248865911077
// MI455X (gfx1250) — matrix operand order repaired and checked by reading, not compiled, not run on hardware
//
#include <hip/hip_runtime.h>

#define VOCAB 50257
#define EMBED 512
#define CTXD  512
#define HID   1024
#define NLAY  6
#define BATCH 2
#define SEQL  2048
#define ROWS  (BATCH * SEQL)   /* 4096 */

typedef __attribute__((ext_vector_type(16))) __bf16 v16bf;
typedef __attribute__((ext_vector_type(8)))  float  v8f;
typedef int v4i __attribute__((vector_size(16)));

#define KC   128
#define TM   32
#define TN   128
#define LPAD 8                       /* row stride 136 elems = 272B */
#define KW   (KC + LPAD)             /* 136 */
#define NT   ((VOCAB + TN - 1) / TN) /* 393 column tiles */
#define ATILE (2 * TM * KW)          /* 8704 bf16  = 17408 B per (rb,kc) */
#define BTILE (2 * TN * KW)          /* 34816 bf16 = 69632 B per (nt,kc) */

// ---- gfx1250 async global->LDS copy (fallback: plain 16B copy) -------------
#if defined(__has_builtin)
#if __has_builtin(__builtin_amdgcn_global_load_async_to_lds_b128) && \
    __has_builtin(__builtin_amdgcn_s_wait_asynccnt)
#define USE_ASYNC_LDS 1
#endif
#endif

__device__ __forceinline__ void cp16_g2l(const void* g, void* l) {
#ifdef USE_ASYNC_LDS
    __builtin_amdgcn_global_load_async_to_lds_b128((v4i*)g, (v4i*)l, 0, 0);
#else
    *(float4*)l = *(const float4*)g;
#endif
}
__device__ __forceinline__ void cp_wait() {
#ifdef USE_ASYNC_LDS
    __builtin_amdgcn_s_wait_asynccnt(0);
#endif
}

// ---------------------------------------------------------------------------
// Kernel 1: tok = LayerNorm(embedding[input_ids])  (one wave32 per row)
// ---------------------------------------------------------------------------
__global__ void embed_ln_kernel(const int* __restrict__ ids,
                                const float* __restrict__ emb,
                                const float* __restrict__ gamma,
                                const float* __restrict__ beta,
                                float* __restrict__ tok) {
    const int wave = threadIdx.x >> 5;
    const int lane = threadIdx.x & 31;
    const int row  = blockIdx.x * 8 + wave;
    if (row >= ROWS) return;
    const int id = ids[row];
    const float* e = emb + (long long)id * EMBED;

    float x[16];
    float s = 0.f;
#pragma unroll
    for (int i = 0; i < 16; ++i) { x[i] = e[lane + 32 * i]; s += x[i]; }
#pragma unroll
    for (int m = 16; m >= 1; m >>= 1) s += __shfl_xor(s, m, 32);
    const float mu = s * (1.0f / EMBED);

    float v = 0.f;
#pragma unroll
    for (int i = 0; i < 16; ++i) { float d = x[i] - mu; v += d * d; }
#pragma unroll
    for (int m = 16; m >= 1; m >>= 1) v += __shfl_xor(v, m, 32);
    const float rstd = rsqrtf(v * (1.0f / EMBED) + 1e-5f);

    float* o = tok + (long long)row * EMBED;
#pragma unroll
    for (int i = 0; i < 16; ++i) {
        const int c = lane + 32 * i;
        o[c] = gamma[c] * (x[i] - mu) * rstd + beta[c];
    }
}

// ---------------------------------------------------------------------------
// Kernel 2: sequential CVFP recurrence, one persistent 1024-thread workgroup.
// ---------------------------------------------------------------------------
__global__ void __launch_bounds__(1024, 1)
recurrence_kernel(const float* __restrict__ tok,
                  const float* __restrict__ Wc, const float* __restrict__ bc,
                  const float* __restrict__ Wt, const float* __restrict__ bt,
                  float* __restrict__ ctxs) {
    __shared__ float h0[HID];
    __shared__ float h1[HID];
    const int  j   = threadIdx.x;
    const bool isC = (j < CTXD);
    const float* Wbase = isC ? (Wc + j) : (Wt + (j - CTXD));
    const float* bbase = isC ? (bc + j) : (bt + (j - CTXD));

    h0[j] = 0.f;
    h1[j] = 0.f;
    __syncthreads();

    for (int s = 0; s < SEQL; ++s) {
        if (!isC) {
            h0[j] = tok[(long long)(0 * SEQL + s) * EMBED + (j - CTXD)];
            h1[j] = tok[(long long)(1 * SEQL + s) * EMBED + (j - CTXD)];
        }
        __syncthreads();

        for (int l = 0; l < NLAY; ++l) {
            const float* Wl = Wbase + (long long)l * HID * CTXD;
            float a0 = 0.f, a1 = 0.f;
#pragma unroll 4
            for (int k = 0; k < HID; ++k) {
                const float w = Wl[(long long)k * CTXD];
                a0 = fmaf(h0[k], w, a0);
                a1 = fmaf(h1[k], w, a1);
            }
            const float bl = bbase[l * CTXD];
            const float n0 = h0[j] + tanhf(a0 + bl);
            const float n1 = h1[j] + tanhf(a1 + bl);
            __syncthreads();
            h0[j] = n0;
            h1[j] = n1;
            __syncthreads();
        }
        if (isC) {
            ctxs[(long long)(0 * SEQL + s) * CTXD + j] = h0[j];
            ctxs[(long long)(1 * SEQL + s) * CTXD + j] = h1[j];
        }
    }
}

// ---------------------------------------------------------------------------
// Kernel 3a: pre-split A (=ctxs) into bf16 hi/lo tiles [rb][kc][2][TM][KW],
// identical to the GEMM's LDS image (row-major, k-contiguous, padded).
// ---------------------------------------------------------------------------
__global__ void split_a_kernel(const float* __restrict__ A, __bf16* __restrict__ gA) {
    const int rb = blockIdx.x, kc = blockIdx.y;
    __bf16* t = gA + ((size_t)rb * 4 + kc) * ATILE;
    for (int idx = threadIdx.x; idx < TM * KW; idx += 256) {
        const int m = idx / KW, k = idx % KW;
        float x = 0.f;
        if (k < KC) x = A[(long long)(rb * TM + m) * 512 + kc * KC + k];
        const __bf16 h = (__bf16)x;
        t[idx]           = h;                       // hi plane
        t[TM * KW + idx] = (__bf16)(x - (float)h);  // lo plane
    }
}

// ---------------------------------------------------------------------------
// Kernel 3b: pre-split W_out into TRANSPOSED bf16 hi/lo tiles
// [nt][kc][2][TN][KW]  (LDS image: [n][k], k-contiguous per lane).
// Transpose goes through LDS so global reads AND writes stay coalesced.
// ---------------------------------------------------------------------------
__global__ void split_b_kernel(const float* __restrict__ Bm, __bf16* __restrict__ gB) {
    __shared__ __align__(16) __bf16 T[2][TN][KW];   // 69.6 KB
    const int nt = blockIdx.x, kc = blockIdx.y;
    const int colBase = nt * TN, kBase = kc * KC;

    for (int idx = threadIdx.x; idx < 2 * TN * KW; idx += 256)
        (&T[0][0][0])[idx] = (__bf16)0.f;           // deterministic pads
    __syncthreads();

    for (int idx = threadIdx.x; idx < KC * TN; idx += 256) {
        const int r = idx >> 7;          // k
        const int c = idx & (TN - 1);    // n
        const int col = colBase + c;
        const float x = (col < VOCAB) ? Bm[(long long)(kBase + r) * VOCAB + col] : 0.f;
        const __bf16 h = (__bf16)x;
        T[0][c][r] = h;
        T[1][c][r] = (__bf16)(x - (float)h);
    }
    __syncthreads();

    __bf16* t = gB + ((size_t)nt * 4 + kc) * BTILE;
    const float4* src = (const float4*)&T[0][0][0];
    float4* dst = (float4*)t;
    for (int idx = threadIdx.x; idx < BTILE / 8; idx += 256)  // 16B chunks
        dst[idx] = src[idx];
}

// ---------------------------------------------------------------------------
// Kernel 3c: logits = ctxs @ W_out + b_out via v_wmma_f32_16x16x32_bf16,
// 2-term bf16 split (AhBh + AhBl + AlBh ~= fp32). Staging is a pure flat
// async global->LDS copy of pre-split tiles; hot loop = ds_load_b128 + wmma.
// ---------------------------------------------------------------------------
__global__ void __launch_bounds__(256, 2)
gemm_kernel(const __bf16* __restrict__ gA,   // [128][4][2][TM][KW]
            const __bf16* __restrict__ gB,   // [393][4][2][TN][KW]
            const float* __restrict__ bias,
            float* __restrict__ out) {
    __shared__ __align__(16) __bf16 As[2][TM][KW];  // 17.4 KB
    __shared__ __align__(16) __bf16 Bs[2][TN][KW];  // 69.6 KB

    const int tid  = threadIdx.x;
    const int lane = tid & 31;
    const int wave = tid >> 5;
    const int wm   = wave >> 2;   // 0..1
    const int wn   = wave & 3;    // 0..3 (32-col slab = two 16x16 tiles)
    const int rowBase = blockIdx.y * TM;
    const int colBase = blockIdx.x * TN;

    v8f acc0 = {0.f, 0.f, 0.f, 0.f, 0.f, 0.f, 0.f, 0.f};
    v8f acc1 = {0.f, 0.f, 0.f, 0.f, 0.f, 0.f, 0.f, 0.f};

    const int mrow = (lane & 15) + wm * 16;
    const int n0   = (lane & 15) + wn * 32;
    const int kbA  = (lane < 16) ? 0 : 16;   // contiguous k = 16h + s (matches B)
    const int kbB  = (lane < 16) ? 0 : 16;   // contiguous k = 16h + s (matches A)

    for (int kc = 0; kc < 4; ++kc) {
        // ---- stage pre-split tiles: flat 16B async copies ----
        {
            const char* ga = (const char*)(gA + ((size_t)blockIdx.y * 4 + kc) * ATILE);
            char*       la = (char*)&As[0][0][0];
            for (int off = tid * 16; off < ATILE * 2; off += 256 * 16)
                cp16_g2l(ga + off, la + off);
            const char* gb = (const char*)(gB + ((size_t)blockIdx.x * 4 + kc) * BTILE);
            char*       lb = (char*)&Bs[0][0][0];
            for (int off = tid * 16; off < BTILE * 2; off += 256 * 16)
                cp16_g2l(gb + off, lb + off);
        }
        cp_wait();
        __syncthreads();

#pragma unroll
        for (int kk = 0; kk < KC; kk += 32) {
            v16bf ah, al;
            {
                const __bf16* pah = &As[0][mrow][kk + kbA];
                const __bf16* pal = &As[1][mrow][kk + kbA];
#pragma unroll
                for (int e = 0; e < 8; ++e) {
                    ah[e]     = pah[e];
                    ah[e + 8] = pah[e + 8];
                    al[e]     = pal[e];
                    al[e + 8] = pal[e + 8];
                }
            }
            v16bf bh0, bl0, bh1, bl1;
            {
                const __bf16* pbh0 = &Bs[0][n0][kk + kbB];
                const __bf16* pbl0 = &Bs[1][n0][kk + kbB];
                const __bf16* pbh1 = &Bs[0][n0 + 16][kk + kbB];
                const __bf16* pbl1 = &Bs[1][n0 + 16][kk + kbB];
#pragma unroll
                for (int e = 0; e < 16; ++e) {
                    bh0[e] = pbh0[e];
                    bl0[e] = pbl0[e];
                    bh1[e] = pbh1[e];
                    bl1[e] = pbl1[e];
                }
            }
            acc0 = __builtin_amdgcn_wmma_f32_16x16x32_bf16(
                false, ah, false, bh0, (short)0, acc0, false, false);
            acc0 = __builtin_amdgcn_wmma_f32_16x16x32_bf16(
                false, ah, false, bl0, (short)0, acc0, false, false);
            acc0 = __builtin_amdgcn_wmma_f32_16x16x32_bf16(
                false, al, false, bh0, (short)0, acc0, false, false);
            acc1 = __builtin_amdgcn_wmma_f32_16x16x32_bf16(
                false, ah, false, bh1, (short)0, acc1, false, false);
            acc1 = __builtin_amdgcn_wmma_f32_16x16x32_bf16(
                false, ah, false, bl1, (short)0, acc1, false, false);
            acc1 = __builtin_amdgcn_wmma_f32_16x16x32_bf16(
                false, al, false, bh1, (short)0, acc1, false, false);
        }
        __syncthreads();
    }

    // ---- store per ISA C/D layout: VGPR r -> M=r (lanes<16) / r+8 (>=16) ----
    const int rbase = rowBase + wm * 16 + ((lane < 16) ? 0 : 8);
    const int col0  = colBase + wn * 32 + (lane & 15);
    const int col1  = col0 + 16;
    if (col0 < VOCAB) {
        const float bv = bias[col0];
#pragma unroll
        for (int r = 0; r < 8; ++r)
            out[(long long)(rbase + r) * VOCAB + col0] = acc0[r] + bv;
    }
    if (col1 < VOCAB) {
        const float bv = bias[col1];
#pragma unroll
        for (int r = 0; r < 8; ++r)
            out[(long long)(rbase + r) * VOCAB + col1] = acc1[r] + bv;
    }
}

// ---------------------------------------------------------------------------
extern "C" void kernel_launch(void* const* d_in, const int* in_sizes, int n_in,
                              void* d_out, int out_size, void* d_ws, size_t ws_size,
                              hipStream_t stream) {
    const int*   ids   = (const int*)d_in[0];
    const float* emb   = (const float*)d_in[1];
    const float* gamma = (const float*)d_in[2];
    const float* beta  = (const float*)d_in[3];
    const float* Wc    = (const float*)d_in[4];
    const float* bc    = (const float*)d_in[5];
    const float* Wt    = (const float*)d_in[6];
    const float* bt    = (const float*)d_in[7];
    const float* Wout  = (const float*)d_in[8];
    const float* bout  = (const float*)d_in[9];
    float* out = (float*)d_out;

    float*  tok  = (float*)d_ws;                        // 8 MB
    float*  ctxs = tok + (size_t)ROWS * EMBED;          // 8 MB
    __bf16* gA   = (__bf16*)(ctxs + (size_t)ROWS * CTXD);        // 8.9 MB
    __bf16* gB   = gA + (size_t)(ROWS / TM) * 4 * ATILE;         // 109.4 MB

    embed_ln_kernel<<<ROWS / 8, 256, 0, stream>>>(ids, emb, gamma, beta, tok);
    recurrence_kernel<<<1, 1024, 0, stream>>>(tok, Wc, bc, Wt, bt, ctxs);
    split_a_kernel<<<dim3(ROWS / TM, 4), 256, 0, stream>>>(ctxs, gA);
    split_b_kernel<<<dim3(NT, 4), 256, 0, stream>>>(Wout, gB);
    gemm_kernel<<<dim3(NT, ROWS / TM), 256, 0, stream>>>(gA, gB, bout, out);
}
